// ActorSNN_76682346103358
// MI455X (gfx1250) — matrix operand order repaired and checked by reading, not compiled, not run on hardware
//
#include <hip/hip_runtime.h>
#include <math.h>

#define B_ROWS  8192
#define S_DIM   256
#define H_DIM   1024
#define M_TILE  32
#define T_STEPS 8

typedef _Float16 v16h __attribute__((ext_vector_type(16)));
typedef float    v8f  __attribute__((ext_vector_type(8)));

union FragU {
  uint32_t w[8];
  uint4    q[2];
  v16h     h;
};

// ---------------------------------------------------------------------------
// fp32 -> fp16 conversion (prep for WMMA operands)
// ---------------------------------------------------------------------------
__global__ void cvt_f32_to_f16(const float* __restrict__ src,
                               _Float16* __restrict__ dst, int n) {
  int i = blockIdx.x * blockDim.x + threadIdx.x;
  int stride = gridDim.x * blockDim.x;
  for (; i < n; i += stride) dst[i] = (_Float16)src[i];
}

// A-fragment (16x32 f16) from row-major global, per ISA A layout:
// lane L: row = L%16; v0..3 = K (sel..sel+7), v4..7 = K (sel+16..sel+23), sel = (L<16?0:8)
__device__ __forceinline__ v16h load_a_global(const _Float16* __restrict__ base,
                                              int row, int ld, int k0, int lhalf) {
  FragU f;
  const _Float16* p = base + (size_t)row * ld + k0 + lhalf * 8;
  f.q[0] = *(const uint4*)(p);
  f.q[1] = *(const uint4*)(p + 16);
  return f.h;
}

// B-fragment (32x16 f16) where B(k,j) = W[j,k] (W row-major):
// lane L: col j = jblk*16 + L%16; K = k0 + sel..sel+7, sel+16..sel+23, sel = (L<16?0:8)
__device__ __forceinline__ v16h load_b_global(const _Float16* __restrict__ w,
                                              int col, int ld, int k0, int lhalf) {
  FragU f;
  const _Float16* p = w + (size_t)col * ld + k0 + lhalf * 8;
  f.q[0] = *(const uint4*)(p);
  f.q[1] = *(const uint4*)(p + 16);
  return f.h;
}

// A-fragment for timestep t unpacked from the LDS spike-bit row.
// p points at bits[row] + sel (sel folded in by caller); spike bit t of each
// byte -> f16 0/1. v_perm_b32 gathers bytes (b_even -> byte0, b_odd -> byte2,
// zeros elsewhere), then *0x3C00 turns each 0/1 into packed f16 {0|1, 0|1}.
__device__ __forceinline__ v16h unpack_a_bits(const unsigned char* p, int t) {
  uint2 a0 = *(const uint2*)(p);       // K sel..sel+7
  uint2 a1 = *(const uint2*)(p + 16);  // K sel+16..sel+23
  FragU f;
  uint32_t s;
  s = (a0.x >> t) & 0x01010101u;
  f.w[0] = __builtin_amdgcn_perm(s, s, 0x0C010C00u) * 0x3C00u;
  f.w[1] = __builtin_amdgcn_perm(s, s, 0x0C030C02u) * 0x3C00u;
  s = (a0.y >> t) & 0x01010101u;
  f.w[2] = __builtin_amdgcn_perm(s, s, 0x0C010C00u) * 0x3C00u;
  f.w[3] = __builtin_amdgcn_perm(s, s, 0x0C030C02u) * 0x3C00u;
  s = (a1.x >> t) & 0x01010101u;
  f.w[4] = __builtin_amdgcn_perm(s, s, 0x0C010C00u) * 0x3C00u;
  f.w[5] = __builtin_amdgcn_perm(s, s, 0x0C030C02u) * 0x3C00u;
  s = (a1.y >> t) & 0x01010101u;
  f.w[6] = __builtin_amdgcn_perm(s, s, 0x0C010C00u) * 0x3C00u;
  f.w[7] = __builtin_amdgcn_perm(s, s, 0x0C030C02u) * 0x3C00u;
  return f.h;
}

// ---------------------------------------------------------------------------
// Fused SNN kernel: one workgroup (8 wave32s) per 32 batch rows.
// ---------------------------------------------------------------------------
__global__ __launch_bounds__(256) void snn_fused(
    const _Float16* __restrict__ st16,   // [8192,256]  f16 state
    const _Float16* __restrict__ win16,  // [1024,256]  f16 W_in
    const _Float16* __restrict__ wh16,   // [1024,1024] f16 W_h
    const float* __restrict__ b_in, const float* __restrict__ beta_in,
    const float* __restrict__ th_in, const float* __restrict__ b_h,
    const float* __restrict__ beta_h, const float* __restrict__ th_h,
    const float* __restrict__ w_out, const float* __restrict__ b_out,
    float* __restrict__ out) {
  __shared__ unsigned char lds_bits[M_TILE * H_DIM];  // 32 KB: 8 spike bits / element
  __shared__ float lds_rate[M_TILE];

  const int tid   = threadIdx.x;
  const int lane  = tid & 31;
  const int wave  = tid >> 5;       // 0..7
  const int l16   = lane & 15;
  const int lhalf = lane >> 4;      // 0/1
  const int row0  = blockIdx.x * M_TILE;

  if (tid < M_TILE) lds_rate[tid] = 0.0f;

  v8f vzero = {};

  // ---- Phase 1: x_in tile = state @ W_in^T (+b_in); LIF-in 8 steps -> bit matrix ----
  #pragma unroll 1
  for (int ci = 0; ci < 8; ++ci) {
    const int cb  = wave * 8 + ci;   // 16-wide column block, 0..63
    const int col = cb * 16 + l16;   // this lane's H column (C/D layout: N = lane%16)
    v8f acc[2]; acc[0] = vzero; acc[1] = vzero;
    #pragma unroll 1
    for (int k0 = 0; k0 < S_DIM; k0 += 32) {
      v16h bf = load_b_global(win16, col, S_DIM, k0, lhalf);
      v16h a0 = load_a_global(st16, row0 + l16,      S_DIM, k0, lhalf);
      acc[0] = __builtin_amdgcn_wmma_f32_16x16x32_f16(false, a0, false, bf,
                                                      (short)0, acc[0], false, false);
      v16h a1 = load_a_global(st16, row0 + 16 + l16, S_DIM, k0, lhalf);
      acc[1] = __builtin_amdgcn_wmma_f32_16x16x32_f16(false, a1, false, bf,
                                                      (short)0, acc[1], false, false);
    }
    const float binv  = b_in[col];
    const float betav = fminf(fmaxf(beta_in[col], 0.0f), 1.0f);
    const float thv   = th_in[col];
    #pragma unroll
    for (int rb = 0; rb < 2; ++rb) {
      #pragma unroll
      for (int i = 0; i < 8; ++i) {
        float x   = acc[rb][i] + binv;   // x_in, constant over timesteps
        float mem = 0.0f;
        unsigned bits = 0u;
        #pragma unroll
        for (int t = 0; t < T_STEPS; ++t) {
          float reset = (mem - thv > 0.0f) ? thv : 0.0f;  // detached reset
          mem = betav * mem + x - reset;
          bits |= ((mem - thv > 0.0f) ? 1u : 0u) << t;    // spk1[t]
        }
        int mrow = rb * 16 + lhalf * 8 + i;  // C-layout: VGPR i -> M = i (+8 for hi half)
        lds_bits[mrow * H_DIM + col] = (unsigned char)bits;
      }
    }
  }
  __syncthreads();

  // ---- Phase 2: each wave owns ONE row-block and 16 column blocks so that a
  //      single unpacked A-fragment feeds 8 WMMAs per K-iteration. The W_out
  //      dot product is folded into the timestep loop (pacc), eliminating the
  //      per-column spike-sum registers. ----
  const int rb2     = wave & 1;            // row block 0/1
  const int cb2base = (wave >> 1) * 16;    // 16 column blocks per wave
  // fold the lane-constant half-wave selector into the base pointer
  const unsigned char* bitrow =
      lds_bits + (rb2 * 16 + l16) * H_DIM + lhalf * 8;

  #pragma unroll 1
  for (int g = 0; g < 2; ++g) {            // two groups of 8 column blocks
    const _Float16* bptr[8];
    float bhv[8], betav[8], thv[8], wov[8];
    #pragma unroll
    for (int c = 0; c < 8; ++c) {
      int col  = (cb2base + g * 8 + c) * 16 + l16;
      bptr[c]  = wh16 + (size_t)col * H_DIM + lhalf * 8;  // lane-constant folded
      bhv[c]   = b_h[col];
      betav[c] = fminf(fmaxf(beta_h[col], 0.0f), 1.0f);
      thv[c]   = th_h[col];
      wov[c]   = w_out[col] * 0.125f;      // rate = ssum/8 folded in
    }
    v8f mem[8];
    #pragma unroll
    for (int c = 0; c < 8; ++c) mem[c] = vzero;
    v8f pacc = vzero;                      // running rate . W_out partials

    #pragma unroll 1
    for (int t = 0; t < T_STEPS; ++t) {
      v8f acc[8];
      #pragma unroll
      for (int c = 0; c < 8; ++c) acc[c] = vzero;
      #pragma unroll 2
      for (int k0 = 0; k0 < H_DIM; k0 += 32) {
        v16h af = unpack_a_bits(bitrow + k0, t);
        #pragma unroll
        for (int c = 0; c < 8; ++c) {
          FragU bf;
          const _Float16* p = bptr[c] + k0;
          bf.q[0] = *(const uint4*)(p);
          bf.q[1] = *(const uint4*)(p + 16);
          acc[c] = __builtin_amdgcn_wmma_f32_16x16x32_f16(false, af, false, bf.h,
                                                          (short)0, acc[c], false, false);
        }
      }
      // LIF-h update + fold spk2 straight into the W_out dot product
      #pragma unroll
      for (int c = 0; c < 8; ++c) {
        #pragma unroll
        for (int i = 0; i < 8; ++i) {
          float h = acc[c][i] + bhv[c];
          float m = mem[c][i];
          float r = (m - thv[c] > 0.0f) ? thv[c] : 0.0f;   // detached reset
          m = betav[c] * m + h - r;
          mem[c][i] = m;
          pacc[i] += (m - thv[c] > 0.0f) ? wov[c] : 0.0f;  // spk2 * W_out/8
        }
      }
    }
    // reduce pacc across the 16 lanes of each half-wave (same row, 16 cols)
    #pragma unroll
    for (int i = 0; i < 8; ++i) {
      float v = pacc[i];
      #pragma unroll
      for (int msk = 1; msk < 16; msk <<= 1) v += __shfl_xor(v, msk, 32);
      if (l16 == 0) atomicAdd(&lds_rate[rb2 * 16 + lhalf * 8 + i], v);
    }
  }
  __syncthreads();

  if (tid < M_TILE) {
    out[row0 + tid] = tanhf(lds_rate[tid] + b_out[0]);
  }
}

// ---------------------------------------------------------------------------
extern "C" void kernel_launch(void* const* d_in, const int* in_sizes, int n_in,
                              void* d_out, int out_size, void* d_ws, size_t ws_size,
                              hipStream_t stream) {
  const float* state   = (const float*)d_in[0];
  const float* W_in    = (const float*)d_in[1];
  const float* b_in    = (const float*)d_in[2];
  const float* beta_in = (const float*)d_in[3];
  const float* th_in   = (const float*)d_in[4];
  const float* W_h     = (const float*)d_in[5];
  const float* b_h     = (const float*)d_in[6];
  const float* beta_h  = (const float*)d_in[7];
  const float* th_h    = (const float*)d_in[8];
  const float* W_out   = (const float*)d_in[9];
  const float* b_out   = (const float*)d_in[10];
  float* out = (float*)d_out;

  char* ws = (char*)d_ws;
  size_t off = 0;
  _Float16* st16  = (_Float16*)(ws + off); off += (size_t)B_ROWS * S_DIM * sizeof(_Float16);
  _Float16* win16 = (_Float16*)(ws + off); off += (size_t)H_DIM * S_DIM * sizeof(_Float16);
  _Float16* wh16  = (_Float16*)(ws + off); off += (size_t)H_DIM * H_DIM * sizeof(_Float16);
  (void)ws_size; (void)in_sizes; (void)n_in; (void)out_size;

  cvt_f32_to_f16<<<2048, 256, 0, stream>>>(state, st16, B_ROWS * S_DIM);
  cvt_f32_to_f16<<<256,  256, 0, stream>>>(W_in,  win16, H_DIM * S_DIM);
  cvt_f32_to_f16<<<1024, 256, 0, stream>>>(W_h,   wh16,  H_DIM * H_DIM);

  snn_fused<<<B_ROWS / M_TILE, 256, 0, stream>>>(
      st16, win16, wh16, b_in, beta_in, th_in, b_h, beta_h, th_h,
      W_out, b_out, out);
}
